// StandardAttention_28939489640870
// MI455X (gfx1250) — matrix operand order repaired and checked by reading, not compiled, not run on hardware
//
#include <hip/hip_runtime.h>
#include <hip/hip_bf16.h>
#include <cstdint>

// ---------------------------------------------------------------------------
// MI455X (gfx1250) multi-head attention: f16 WMMA + Tensor Data Mover staging.
//   H=2048, NH=32, HD=64, B=2, S=2048, M=B*S=4096.
// ---------------------------------------------------------------------------

typedef __attribute__((ext_vector_type(16))) _Float16 v16h;
typedef __attribute__((ext_vector_type(8)))  float    v8f;
typedef __attribute__((ext_vector_type(4)))  unsigned int u32x4;
typedef __attribute__((ext_vector_type(8)))  int      i32x8;
typedef __attribute__((ext_vector_type(4)))  int      i32x4;

union Frag {
    v16h h;
    uint4 u[2];
};

__device__ __forceinline__ v8f wmma16(v16h a, v16h b, v8f c) {
    // D = A(16x32 f16) * B(32x16 f16) + C(16x16 f32)
    return __builtin_amdgcn_wmma_f32_16x16x32_f16(
        /*neg_a=*/false, a, /*neg_b=*/false, b,
        /*c_mod=*/(short)0, c, /*reuse_a=*/false, /*reuse_b=*/false);
}

#define HDIM   2048
#define MROWS  4096
#define NHEADS 32
#define HEADD  64
#define SEQ    2048

#if defined(__gfx1250__) && __has_builtin(__builtin_amdgcn_tensor_load_to_lds) && \
    __has_builtin(__builtin_amdgcn_s_wait_tensorcnt)
#define USE_TDM 1
#endif

#ifdef USE_TDM
// ---------------------------------------------------------------------------
// Issue one TDM 2D tile load: 32 (K, f16) x 128 (rows) from a K-major matrix
// (row pitch 2048 f16) into LDS at byte offset lds_off, with TDM padding of
// 4 DWORDs (16B) after every 16 DWORDs (one 64B row) -> LDS pitch 40 f16.
// D# packing per CDNA5 ISA 8.3/8.4 (group0 128b, group1 256b, groups 2/3 = 0).
// This toolchain's builtin is the 6-arg form:
//   (u32x4 g0, i32x8 g1, i32x4 g2, i32x4 g3, i32x8, i32 cpol)
// ---------------------------------------------------------------------------
__device__ __forceinline__ void tdm_load_tile_128x32(const _Float16* gsrc,
                                                     unsigned lds_off,
                                                     unsigned tensor_rows) {
    unsigned long long ga = (unsigned long long)(uintptr_t)gsrc; // flat==global VA
    u32x4 g0;
    g0[0] = 1u;                                            // count=1, user desc
    g0[1] = lds_off;                                       // lds_addr (bytes)
    g0[2] = (unsigned)ga;                                  // global_addr[31:0]
    g0[3] = (unsigned)((ga >> 32) & 0x01FFFFFFu)           // global_addr[56:32]
          | (2u << 30);                                    // type = 2 ("image")
    i32x8 g1;
    g1[0] = (int)((1u << 16)       // data_size = 1 -> 2 bytes
                | (1u << 20)       // pad_enable (load only)
                | (3u << 22)       // pad_interval code 3 -> every 16 DWORDs
                | (3u << 25));     // pad_amount  code 3 -> 4 DWORDs (16B)
    g1[1] = (int)(((unsigned)HDIM & 0xFFFFu) << 16);       // tensor_dim0[15:0]
    g1[2] = (int)((((unsigned)HDIM >> 16) & 0xFFFFu)       // tensor_dim0[31:16]
                | ((tensor_rows & 0xFFFFu) << 16));        // tensor_dim1[15:0]
    g1[3] = (int)(((tensor_rows >> 16) & 0xFFFFu)          // tensor_dim1[31:16]
                | (32u << 16));                            // tile_dim0 = 32
    g1[4] = (int)(128u);                                   // tile_dim1=128, tile_dim2=0
    g1[5] = (int)((unsigned)HDIM);                         // tensor_dim0_stride[31:0]
    g1[6] = 0;                                             // stride0 hi / stride1 lo
    g1[7] = 0;                                             // stride1 hi
    const i32x4 gz4 = (i32x4){0, 0, 0, 0};                 // groups 2/3 unused (2D)
    const i32x8 gz8 = (i32x8){0, 0, 0, 0, 0, 0, 0, 0};
    __builtin_amdgcn_tensor_load_to_lds(g0, g1, gz4, gz4, gz8, 0);
}
#endif // USE_TDM

// ---------------------------------------------------------------------------
// fp32 -> f16 conversion (vectorized x4)
// ---------------------------------------------------------------------------
__global__ __launch_bounds__(256) void cvt_f32_to_f16(const float* __restrict__ in,
                                                      _Float16* __restrict__ out,
                                                      int n4) {
    int i = blockIdx.x * 256 + threadIdx.x;
    if (i < n4) {
        float4 f = ((const float4*)in)[i];
        union { _Float16 h[4]; uint2 u; } r;
        r.h[0] = (_Float16)f.x; r.h[1] = (_Float16)f.y;
        r.h[2] = (_Float16)f.z; r.h[3] = (_Float16)f.w;
        ((uint2*)out)[i] = r.u;
    }
}

// ---------------------------------------------------------------------------
// GEMM: C[M,N] = A[M,K] * W[N,K]^T + bias[N],  M=4096, N=2048, K=2048, f16 in,
// f32 accumulate. Block: 256 threads (8 waves), 128x128 tile, K-step 32.
// Wave (wm 0..3, wn 0..1) owns a 32x64 sub-tile = 2x4 WMMA fragments.
// Staging: TDM double-buffered (wave0 -> A tile, wave1 -> W tile), one tile in
// flight, s_wait_tensorcnt + barrier. Fallback: direct global->LDS loads.
// STORE_MODE: 0 = f16 [M,N]; 1 = f16 transposed-per-head [B,NH,HD,S]; 2 = f32 [M,N]
// ---------------------------------------------------------------------------
template <int STORE_MODE>
__global__ __launch_bounds__(256) void gemm_wmma_f16(const _Float16* __restrict__ A,
                                                     const _Float16* __restrict__ W,
                                                     const float* __restrict__ bias,
                                                     void* __restrict__ outp) {
    __shared__ _Float16 As[2][128 * 40];
    __shared__ _Float16 Bs[2][128 * 40];

    const int tid  = threadIdx.x;
    const int lane = tid & 31;
    const int hi   = lane >> 4;     // lane half (0/1)
    const int l15  = lane & 15;
    const int wave = tid >> 5;      // 0..7
    const int wm   = wave >> 1;     // 0..3  (rows)
    const int wn   = wave & 1;      // 0..1  (cols)

    const int m0 = blockIdx.y * 128;
    const int n0 = blockIdx.x * 128;

    v8f acc[2][4];
#pragma unroll
    for (int mi = 0; mi < 2; ++mi)
#pragma unroll
        for (int ni = 0; ni < 4; ++ni)
            acc[mi][ni] = (v8f){};

    const int NT = HDIM / 32;

#ifdef USE_TDM
    // ---- prologue: DMA tile 0 into buffer 0 ----
    if (wave == 0)
        tdm_load_tile_128x32(A + (size_t)m0 * HDIM, (unsigned)(uintptr_t)&As[0][0], MROWS);
    else if (wave == 1)
        tdm_load_tile_128x32(W + (size_t)n0 * HDIM, (unsigned)(uintptr_t)&Bs[0][0], HDIM);
#else
    const int r    = tid >> 1;      // 0..127 cooperative-load row
    const int half = tid & 1;       // which 16-f16 half of the 32-wide K slab
#endif

    for (int i = 0; i < NT; ++i) {
        const int kt  = i * 32;
        const int cur = i & 1;

#ifdef USE_TDM
        // ---- software pipeline: DMA tile i+1, wait for tile i, sync ----
        if (i + 1 < NT) {
            const int nxt = (i + 1) & 1;
            if (wave == 0)
                tdm_load_tile_128x32(A + (size_t)m0 * HDIM + kt + 32,
                                     (unsigned)(uintptr_t)&As[nxt][0], MROWS);
            else if (wave == 1)
                tdm_load_tile_128x32(W + (size_t)n0 * HDIM + kt + 32,
                                     (unsigned)(uintptr_t)&Bs[nxt][0], HDIM);
            if (wave < 2) __builtin_amdgcn_s_wait_tensorcnt(1);
        } else {
            if (wave < 2) __builtin_amdgcn_s_wait_tensorcnt(0);
        }
        __syncthreads();
#else
        // ---- cooperative stage 128x32 A tile and 128x32 W tile into LDS ----
        {
            const uint4* gA = (const uint4*)(A + (size_t)(m0 + r) * HDIM + kt + half * 16);
            const uint4* gB = (const uint4*)(W + (size_t)(n0 + r) * HDIM + kt + half * 16);
            uint4 a0 = gA[0], a1 = gA[1];
            uint4 b0 = gB[0], b1 = gB[1];
            *(uint4*)&As[cur][r * 40 + half * 16]     = a0;
            *(uint4*)&As[cur][r * 40 + half * 16 + 8] = a1;
            *(uint4*)&Bs[cur][r * 40 + half * 16]     = b0;
            *(uint4*)&Bs[cur][r * 40 + half * 16 + 8] = b1;
            if (kt + 32 < HDIM) {
                __builtin_prefetch(A + (size_t)(m0 + r) * HDIM + kt + 32 + half * 16, 0, 0);
                __builtin_prefetch(W + (size_t)(n0 + r) * HDIM + kt + 32 + half * 16, 0, 0);
            }
        }
        __syncthreads();
#endif

        // ---- load fragments (ISA 7.12.2 layouts) and multiply ----
        Frag af[2], bf[4];
#pragma unroll
        for (int mi = 0; mi < 2; ++mi) {
            const _Float16* ap = &As[cur][(wm * 32 + mi * 16 + l15) * 40 + hi * 8];
            af[mi].u[0] = *(const uint4*)(ap);
            af[mi].u[1] = *(const uint4*)(ap + 16);
        }
#pragma unroll
        for (int ni = 0; ni < 4; ++ni) {
            const _Float16* bp = &Bs[cur][(wn * 64 + ni * 16 + l15) * 40 + hi * 8];
            bf[ni].u[0] = *(const uint4*)(bp);
            bf[ni].u[1] = *(const uint4*)(bp + 16);
        }
#pragma unroll
        for (int mi = 0; mi < 2; ++mi)
#pragma unroll
            for (int ni = 0; ni < 4; ++ni)
                acc[mi][ni] = wmma16(af[mi].h, bf[ni].h, acc[mi][ni]);
        __syncthreads();
    }

    // ---- epilogue: bias + store. C layout: row = v + 8*hi, col = lane&15 ----
#pragma unroll
    for (int ni = 0; ni < 4; ++ni) {
        const int col = n0 + wn * 64 + ni * 16 + l15;
        const float bv = bias[col];
#pragma unroll
        for (int mi = 0; mi < 2; ++mi) {
#pragma unroll
            for (int v = 0; v < 8; ++v) {
                const int row = m0 + wm * 32 + mi * 16 + v + hi * 8;
                const float val = acc[mi][ni][v] + bv;
                if (STORE_MODE == 0) {
                    ((_Float16*)outp)[(size_t)row * HDIM + col] = (_Float16)val;
                } else if (STORE_MODE == 1) {
                    // V stored transposed per head: [B, NH, HD, S]
                    const int b = row >> 11, s = row & 2047;
                    const int h = col >> 6,  d = col & 63;
                    ((_Float16*)outp)[((size_t)((b * NHEADS + h) * HEADD + d)) * SEQ + s] =
                        (_Float16)val;
                } else {
                    ((float*)outp)[(size_t)row * HDIM + col] = val;
                }
            }
        }
    }
}

// ---------------------------------------------------------------------------
// Causal flash attention, f16 WMMA, f32 softmax state.
// Grid: (S/64, B*NH). Block: 128 threads = 4 waves; wave owns 16 query rows.
// Q:[B,S,H] f16   K:[B,S,H] f16   Vt:[B,NH,HD,S] f16   Oh:[B,S,H] f16
// ---------------------------------------------------------------------------
__global__ __launch_bounds__(128) void attn_wmma(const _Float16* __restrict__ Qh,
                                                 const _Float16* __restrict__ Kh,
                                                 const _Float16* __restrict__ Vt,
                                                 _Float16* __restrict__ Oh) {
    __shared__ _Float16 Plds[4][16 * 72];   // per-wave 16x64 P tile, padded

    const int bh   = blockIdx.y;
    const int b    = bh >> 5;
    const int h    = bh & 31;
    const int qt   = blockIdx.x;            // query tile (64 rows)
    const int wave = threadIdx.x >> 5;
    const int lane = threadIdx.x & 31;
    const int hi   = lane >> 4;
    const int l15  = lane & 15;
    const int qRow0 = qt * 64 + wave * 16;

    // ---- load Q fragments (A-operand): lane l15 = query row, HD split 2x32 ----
    Frag qf[2];
    {
        const _Float16* qp = Qh + ((size_t)(b * SEQ + qRow0 + l15)) * HDIM + h * HEADD;
#pragma unroll
        for (int kk = 0; kk < 2; ++kk) {
            const int c0 = kk * 32 + hi * 8;
            qf[kk].u[0] = *(const uint4*)(qp + c0);
            qf[kk].u[1] = *(const uint4*)(qp + c0 + 16);
        }
    }

    v8f o[4];
#pragma unroll
    for (int nt = 0; nt < 4; ++nt) o[nt] = (v8f){};
    float m_[8], l_[8];
#pragma unroll
    for (int v = 0; v < 8; ++v) { m_[v] = -1e30f; l_[v] = 0.f; }

    const v8f zero = (v8f){};

    for (int j = 0; j <= qt; ++j) {          // causal key-tile bound (uniform)
        const int key0 = j * 64;

        // ---- scores S = Q K^T (16x64 per wave) ----
        v8f s[4];
#pragma unroll
        for (int nt = 0; nt < 4; ++nt) {
            const _Float16* kp =
                Kh + ((size_t)(b * SEQ + key0 + nt * 16 + l15)) * HDIM + h * HEADD;
            Frag k0, k1;
            k0.u[0] = *(const uint4*)(kp + hi * 8);
            k0.u[1] = *(const uint4*)(kp + hi * 8 + 16);
            k1.u[0] = *(const uint4*)(kp + 32 + hi * 8);
            k1.u[1] = *(const uint4*)(kp + 32 + hi * 8 + 16);
            v8f sa = wmma16(qf[0].h, k0.h, zero);
            sa     = wmma16(qf[1].h, k1.h, sa);
            s[nt]  = sa;
        }

        // ---- scale, causal mask, online softmax (rows live in 16-lane halves) ----
#pragma unroll
        for (int v = 0; v < 8; ++v) {
            const int row = qRow0 + v + hi * 8;
            float mx = -1e30f;
#pragma unroll
            for (int nt = 0; nt < 4; ++nt) {
                const int col = key0 + nt * 16 + l15;
                float val = s[nt][v] * 0.125f;            // 1/sqrt(64)
                val = (col <= row) ? val : -1e30f;
                s[nt][v] = val;
                mx = fmaxf(mx, val);
            }
#pragma unroll
            for (int off = 1; off < 16; off <<= 1)
                mx = fmaxf(mx, __shfl_xor(mx, off, 32));  // xor<16 stays in half
            const float nm  = fmaxf(m_[v], mx);
            const float fct = __expf(m_[v] - nm);
            m_[v] = nm;
            float rs = 0.f;
#pragma unroll
            for (int nt = 0; nt < 4; ++nt) {
                const float p = __expf(s[nt][v] - nm);
                s[nt][v] = p;
                rs += p;
                o[nt][v] *= fct;                           // rescale O accum
            }
#pragma unroll
            for (int off = 1; off < 16; off <<= 1)
                rs += __shfl_xor(rs, off, 32);
            l_[v] = l_[v] * fct + rs;
        }

        // ---- redistribute P: C-layout -> A-layout via wave-private LDS ----
#pragma unroll
        for (int v = 0; v < 8; ++v)
#pragma unroll
            for (int nt = 0; nt < 4; ++nt)
                Plds[wave][(v + hi * 8) * 72 + nt * 16 + l15] = (_Float16)s[nt][v];
        asm volatile("s_wait_dscnt 0" ::: "memory");

        Frag pa[2];
#pragma unroll
        for (int kk = 0; kk < 2; ++kk) {
            const _Float16* pp = &Plds[wave][l15 * 72 + kk * 32 + hi * 8];
            pa[kk].u[0] = *(const uint4*)(pp);
            pa[kk].u[1] = *(const uint4*)(pp + 16);
        }

        // ---- O += P * V  (V^T is K-contiguous: [B,NH,HD,S]) ----
#pragma unroll
        for (int nt = 0; nt < 4; ++nt) {
            const _Float16* vp =
                Vt + ((size_t)((b * NHEADS + h) * HEADD + nt * 16 + l15)) * SEQ + key0;
            Frag v0, v1;
            v0.u[0] = *(const uint4*)(vp + hi * 8);
            v0.u[1] = *(const uint4*)(vp + hi * 8 + 16);
            v1.u[0] = *(const uint4*)(vp + 32 + hi * 8);
            v1.u[1] = *(const uint4*)(vp + 32 + hi * 8 + 16);
            o[nt] = wmma16(pa[0].h, v0.h, o[nt]);
            o[nt] = wmma16(pa[1].h, v1.h, o[nt]);
        }
    }

    // ---- normalize and store attention output as f16 [B,S,H] ----
#pragma unroll
    for (int v = 0; v < 8; ++v) {
        const float inv = 1.0f / l_[v];
        const int row = qRow0 + v + hi * 8;
#pragma unroll
        for (int nt = 0; nt < 4; ++nt) {
            Oh[((size_t)(b * SEQ + row)) * HDIM + h * HEADD + nt * 16 + l15] =
                (_Float16)(o[nt][v] * inv);
        }
    }
}

// ---------------------------------------------------------------------------
// Host launcher
// ---------------------------------------------------------------------------
extern "C" void kernel_launch(void* const* d_in, const int* in_sizes, int n_in,
                              void* d_out, int out_size, void* d_ws, size_t ws_size,
                              hipStream_t stream) {
    (void)in_sizes; (void)n_in; (void)out_size; (void)ws_size;

    const float* hidden = (const float*)d_in[0];
    const float* wq = (const float*)d_in[1];
    const float* bq = (const float*)d_in[2];
    const float* wk = (const float*)d_in[3];
    const float* bk = (const float*)d_in[4];
    const float* wv = (const float*)d_in[5];
    const float* bv = (const float*)d_in[6];
    const float* wo = (const float*)d_in[7];
    const float* bo = (const float*)d_in[8];

    char* ws = (char*)d_ws;
    size_t off = 0;
    auto carve = [&](size_t bytes) {
        void* p = ws + off;
        off += (bytes + 255) & ~(size_t)255;
        return p;
    };
    const size_t actB = (size_t)MROWS * HDIM * sizeof(_Float16);   // 16 MB
    const size_t wgtB = (size_t)HDIM * HDIM * sizeof(_Float16);    // 8 MB

    _Float16* Xh  = (_Float16*)carve(actB);
    _Float16* Wqh = (_Float16*)carve(wgtB);
    _Float16* Wkh = (_Float16*)carve(wgtB);
    _Float16* Wvh = (_Float16*)carve(wgtB);
    _Float16* Woh = (_Float16*)carve(wgtB);
    _Float16* Qh  = (_Float16*)carve(actB);
    _Float16* Kh  = (_Float16*)carve(actB);
    _Float16* Vt  = (_Float16*)carve(actB);   // [B, NH, HD, S]
    _Float16* Oh  = (_Float16*)carve(actB);

    // 1) fp32 -> f16 conversions
    {
        const int nAct4 = MROWS * HDIM / 4;
        const int nWgt4 = HDIM * HDIM / 4;
        cvt_f32_to_f16<<<(nAct4 + 255) / 256, 256, 0, stream>>>(hidden, Xh, nAct4);
        cvt_f32_to_f16<<<(nWgt4 + 255) / 256, 256, 0, stream>>>(wq, Wqh, nWgt4);
        cvt_f32_to_f16<<<(nWgt4 + 255) / 256, 256, 0, stream>>>(wk, Wkh, nWgt4);
        cvt_f32_to_f16<<<(nWgt4 + 255) / 256, 256, 0, stream>>>(wv, Wvh, nWgt4);
        cvt_f32_to_f16<<<(nWgt4 + 255) / 256, 256, 0, stream>>>(wo, Woh, nWgt4);
    }

    // 2) projections: grid (N/128, M/128)
    dim3 ggrid(HDIM / 128, MROWS / 128);
    gemm_wmma_f16<0><<<ggrid, 256, 0, stream>>>(Xh, Wqh, bq, (void*)Qh);
    gemm_wmma_f16<0><<<ggrid, 256, 0, stream>>>(Xh, Wkh, bk, (void*)Kh);
    gemm_wmma_f16<1><<<ggrid, 256, 0, stream>>>(Xh, Wvh, bv, (void*)Vt);

    // 3) causal flash attention
    attn_wmma<<<dim3(SEQ / 64, 2 * NHEADS), 128, 0, stream>>>(Qh, Kh, Vt, Oh);

    // 4) output projection -> fp32 d_out
    gemm_wmma_f16<2><<<ggrid, 256, 0, stream>>>(Oh, Woh, bo, d_out);
}
